// TModel_26276609917536
// MI455X (gfx1250) — matrix operand order repaired and checked by reading, not compiled, not run on hardware
//
#include <hip/hip_runtime.h>

// ---------------------------------------------------------------------------
// Graph-net forward for MI455X (gfx1250): bf16 WMMA GEMMs + f32 accumulate,
// TDM (tensor_load_to_lds) double-buffered weight panels in the edge kernel.
// FDIM=128, edge MLP 256->256->256, scatter-sum, node MLP 512->512->128, BN.
// ---------------------------------------------------------------------------

typedef __attribute__((ext_vector_type(8)))  __bf16 v8bf;
typedef __attribute__((ext_vector_type(16))) __bf16 v16bf;
typedef __attribute__((ext_vector_type(8)))  float  v8f;
typedef __attribute__((ext_vector_type(4)))  unsigned int u32x4;
typedef __attribute__((ext_vector_type(8)))  int i32x8;
typedef __attribute__((ext_vector_type(4)))  int i32x4;

union Frag16 { v8bf h[2]; v16bf v; };

#define FD   128
#define FD2  256
#define FD4  512
#define SLOPE 0.1f
#define EPSBN 1e-5f

#define HAVE_TDM __has_builtin(__builtin_amdgcn_tensor_load_to_lds)

__device__ __forceinline__ unsigned short f2bf(float f) {
  unsigned u = __float_as_uint(f);
  unsigned r = u + 0x7FFFu + ((u >> 16) & 1u);   // round-to-nearest-even
  return (unsigned short)(r >> 16);
}

// ---------------------------------------------------------------------------
// TDM: load a [256 rows x 32 elems] bf16 k-panel (tile of a [256 x 256]
// row-major tensor, stride0 = 256 elems) into LDS. gsrc points at element
// [0][k]; LDS receives 256*32 bf16 contiguous (row stride 32 elems).
// ---------------------------------------------------------------------------
__device__ __forceinline__ void tdm_load_panel_256x32(
    const unsigned short* gsrc, unsigned short* lds) {
#if HAVE_TDM
  // Flat LDS pointers keep the LDS byte offset in addr[31:0] (ISA 10.2).
  const unsigned lds_off = (unsigned)(size_t)lds;
  const unsigned long long ga = (unsigned long long)(size_t)gsrc;

  u32x4 g0;
  g0[0] = 1u;                                        // count=1, user mode
  g0[1] = lds_off;                                   // lds_addr [63:32]
  g0[2] = (unsigned)(ga & 0xFFFFFFFFu);              // global_addr lo
  g0[3] = (unsigned)((ga >> 32) & 0x01FFFFFFu)       // global_addr hi (57b)
        | (2u << 30);                                // type = 2 ("image")

  const unsigned dim0 = 256u, dim1 = 256u;           // tensor dims (elems)
  const unsigned tile0 = 32u, tile1 = 256u;          // tile dims
  const unsigned long long s0 = 256ull;              // tensor_dim0_stride
  i32x8 g1 = (i32x8){};
  g1[0] = (int)(1u << 16);                           // data_size=1 -> 2 bytes
  g1[1] = (int)((dim0 & 0xFFFFu) << 16);             // tensor_dim0 @ bit 48
  g1[2] = (int)((dim0 >> 16) | ((dim1 & 0xFFFFu) << 16));
  g1[3] = (int)((dim1 >> 16) | (tile0 << 16));       // tile_dim0 @ bit 112
  g1[4] = (int)tile1;                                // tile_dim1; tile_dim2=0
  g1[5] = (int)(unsigned)(s0 & 0xFFFFFFFFu);         // stride0 lo32
  g1[6] = (int)(unsigned)((s0 >> 32) & 0xFFFFu);     // stride0 hi16
  g1[7] = 0;

  const i32x4 gz = (i32x4){};                        // groups 2/3 unused (2D)
#if defined(__clang_major__) && (__clang_major__ >= 23)
  const i32x8 gz8 = (i32x8){};
  __builtin_amdgcn_tensor_load_to_lds(g0, g1, gz, gz, gz8, 0);
#else
  __builtin_amdgcn_tensor_load_to_lds(g0, g1, gz, gz, 0);
#endif
#else
  // Fallback: single-wave cooperative copy (32 lanes x 16B chunks).
  const int lane = threadIdx.x & 31;
  for (int c = lane; c < (256 * 32) / 8; c += 32) {   // 16B chunks
    const int n = c >> 2, q = c & 3;
    *(v8bf*)(lds + (size_t)c * 8) = *(const v8bf*)(gsrc + (size_t)n * FD2 + q * 8);
  }
#endif
}

__device__ __forceinline__ void tdm_wait() {
#if HAVE_TDM
#if __has_builtin(__builtin_amdgcn_s_wait_tensorcnt)
  __builtin_amdgcn_s_wait_tensorcnt(0);
#else
  asm volatile("s_wait_tensorcnt 0" ::: "memory");
#endif
#endif
}

// ---------- conversion kernels -------------------------------------------------
__global__ void cvt_bf16_kernel(const float* __restrict__ x,
                                unsigned short* __restrict__ o, long n) {
  long i = (long)blockIdx.x * blockDim.x + threadIdx.x;
  if (i < n) o[i] = f2bf(x[i]);
}

// W[din][dout] (f32, row-major) -> Wt[dout][din] (bf16): B-fragments become
// per-lane contiguous 32B loads.
__global__ void cvtT_bf16_kernel(const float* __restrict__ W,
                                 unsigned short* __restrict__ Wt,
                                 int din, int dout) {
  long i = (long)blockIdx.x * blockDim.x + threadIdx.x;
  long total = (long)din * dout;
  if (i >= total) return;
  int k = (int)(i / dout);
  int n = (int)(i % dout);
  Wt[(size_t)n * din + k] = f2bf(W[i]);
}

// ---------- edge MLP: [E,256] -> Linear -> LeakyReLU -> Linear -> scatter-add --
__global__ __launch_bounds__(256)
void edge_mlp_kernel(const unsigned short* __restrict__ xs,     // [S,128] bf16
                     const unsigned short* __restrict__ ea,     // [E,128] bf16
                     const unsigned short* __restrict__ w1aT,   // [256,256] bf16
                     const float* __restrict__ b1a,
                     const unsigned short* __restrict__ w1bT,   // [256,256] bf16
                     const float* __restrict__ b1b,
                     const int* __restrict__ srcI,
                     const int* __restrict__ tgtI,
                     float* __restrict__ agg,                   // [T,256] f32
                     int nE) {
  __shared__ __align__(32) unsigned short h1[64 * FD2];      // 32 KB
  __shared__ __align__(32) unsigned short wpan[2][FD2 * 32]; // 2 x 16 KB

  const int tid  = threadIdx.x;
  const int wv   = tid >> 5;
  const int lane = tid & 31;
  const int mt   = wv & 3;        // m-tile 0..3  (64 rows / 16)
  const int ngrp = wv >> 2;       // 0..1 -> n-tiles [ngrp*8 .. ngrp*8+7]
  const int half = lane >> 4;     // 0: K chunks +0/+8, 1: +16/+24
  const int l16  = lane & 15;
  const int e0   = blockIdx.x * 64;

  // Kick the TDM prefetch of the first W1b k-panel; overlaps all of GEMM1.
  if (wv == 0) tdm_load_panel_256x32(w1bT, &wpan[0][0]);

  // A-fragment row for this lane
  const int rA = mt * 16 + l16;
  int eA = e0 + rA; if (eA > nE - 1) eA = nE - 1;
  const long sA = srcI[eA];
  const unsigned short* rowx = xs + (size_t)sA * FD;
  const unsigned short* rowe = ea + (size_t)eA * FD;
  __builtin_prefetch(rowe, 0, 0);                // global_prefetch_b8

  // ---- GEMM1: msg1 = concat(x_s[src], edge_attr) @ W1a (B direct from L2) ----
  v8f acc[8];
  #pragma unroll
  for (int j = 0; j < 8; ++j) acc[j] = (v8f){};

  for (int k = 0; k < FD2; k += 32) {
    const int k0 = k + half * 16;
    const int k1 = k0 + 8;
    Frag16 a;
    a.h[0] = *(const v8bf*)((k0 < FD) ? (rowx + k0) : (rowe + (k0 - FD)));
    a.h[1] = *(const v8bf*)((k1 < FD) ? (rowx + k1) : (rowe + (k1 - FD)));
    #pragma unroll
    for (int j = 0; j < 8; ++j) {
      const int n = (ngrp * 8 + j) * 16 + l16;
      const v16bf b = *(const v16bf*)(w1aT + (size_t)n * FD2 + k + half * 16);
      acc[j] = __builtin_amdgcn_wmma_f32_16x16x32_bf16(
          false, a.v, false, b, (short)0, acc[j], false, false);
    }
  }

  // bias + LeakyReLU -> LDS (bf16), C-layout: row = mt*16 + half*8 + v
  #pragma unroll
  for (int j = 0; j < 8; ++j) {
    const int col  = (ngrp * 8 + j) * 16 + l16;
    const float bi = b1a[col];
    #pragma unroll
    for (int v = 0; v < 8; ++v) {
      float x = acc[j][v] + bi;
      x = (x >= 0.f) ? x : SLOPE * x;
      h1[(size_t)(mt * 16 + half * 8 + v) * FD2 + col] = f2bf(x);
    }
  }
  if (wv == 0) tdm_wait();        // panel 0 resident
  __syncthreads();                // h1 + panel 0 visible to all waves

  // ---- GEMM2: msg = h1 @ W1b, B from TDM-staged LDS panels (double buffer) ----
  v8f acc2[8];
  #pragma unroll
  for (int j = 0; j < 8; ++j) acc2[j] = (v8f){};

  for (int k = 0; k < FD2; k += 32) {
    const int buf = (k >> 5) & 1;
    if (wv == 0 && (k + 32) < FD2)
      tdm_load_panel_256x32(w1bT + (k + 32), &wpan[buf ^ 1][0]);

    const int k0 = k + half * 16;
    Frag16 a;
    a.h[0] = *(const v8bf*)&h1[(size_t)rA * FD2 + k0];
    a.h[1] = *(const v8bf*)&h1[(size_t)rA * FD2 + k0 + 8];
    #pragma unroll
    for (int j = 0; j < 8; ++j) {
      const int n = (ngrp * 8 + j) * 16 + l16;
      const v16bf b = *(const v16bf*)&wpan[buf][(size_t)n * 32 + half * 16];
      acc2[j] = __builtin_amdgcn_wmma_f32_16x16x32_bf16(
          false, a.v, false, b, (short)0, acc2[j], false, false);
    }
    if (wv == 0) tdm_wait();      // next panel resident before rotation
    __syncthreads();              // all waves done with wpan[buf^1] from i-1
  }

  // ---- scatter-sum into agg via f32 atomics ----
  int trow[8];
  #pragma unroll
  for (int v = 0; v < 8; ++v) {
    const int ee = e0 + mt * 16 + half * 8 + v;
    trow[v] = (ee < nE) ? tgtI[ee] : -1;
  }
  #pragma unroll
  for (int j = 0; j < 8; ++j) {
    const int col  = (ngrp * 8 + j) * 16 + l16;
    const float bi = b1b[col];
    #pragma unroll
    for (int v = 0; v < 8; ++v) {
      if (trow[v] >= 0)
        atomicAdd(agg + (size_t)trow[v] * FD2 + col, acc2[j][v] + bi);
    }
  }
}

// ---------- node MLP: concat(x_t, agg, u) [T,512] -> 512 -> LeakyReLU -> 128 ---
__global__ __launch_bounds__(256)
void node_mlp_kernel(const unsigned short* __restrict__ xt,    // [T,128] bf16
                     const unsigned short* __restrict__ aggb,  // [T,256] bf16
                     const unsigned short* __restrict__ ub,    // [128]   bf16
                     const unsigned short* __restrict__ w2aT,  // [512,512] bf16
                     const float* __restrict__ b2a,
                     const unsigned short* __restrict__ w2bT,  // [128,512] bf16
                     const float* __restrict__ b2b,
                     float* __restrict__ y,                    // [T,128] f32
                     int nT) {
  __shared__ __align__(32) unsigned short h1[32 * FD4];        // 32 KB

  const int tid  = threadIdx.x;
  const int wv   = tid >> 5;
  const int lane = tid & 31;
  const int mt   = wv & 1;        // 32 rows -> 2 m-tiles
  const int ngrp = wv >> 1;       // 0..3
  const int half = lane >> 4;
  const int l16  = lane & 15;
  const int t0   = blockIdx.x * 32;

  const int rA = mt * 16 + l16;
  int t = t0 + rA; if (t > nT - 1) t = nT - 1;
  const unsigned short* rx = xt   + (size_t)t * FD;
  const unsigned short* rg = aggb + (size_t)t * FD2;

  auto chunk = [&](int k0) -> const v8bf* {   // boundaries are multiples of 8
    const unsigned short* p;
    if (k0 < FD)            p = rx + k0;
    else if (k0 < FD + FD2) p = rg + (k0 - FD);
    else                    p = ub + (k0 - FD - FD2);
    return (const v8bf*)p;
  };

  // ---- GEMM1: [32,512] @ W2a [512,512] ----
  v8f acc[8];
  #pragma unroll
  for (int j = 0; j < 8; ++j) acc[j] = (v8f){};

  for (int k = 0; k < FD4; k += 32) {
    const int k0 = k + half * 16;
    Frag16 a;
    a.h[0] = *chunk(k0);
    a.h[1] = *chunk(k0 + 8);
    #pragma unroll
    for (int j = 0; j < 8; ++j) {
      const int n = (ngrp * 8 + j) * 16 + l16;
      const v16bf b = *(const v16bf*)(w2aT + (size_t)n * FD4 + k + half * 16);
      acc[j] = __builtin_amdgcn_wmma_f32_16x16x32_bf16(
          false, a.v, false, b, (short)0, acc[j], false, false);
    }
  }

  #pragma unroll
  for (int j = 0; j < 8; ++j) {
    const int col  = (ngrp * 8 + j) * 16 + l16;
    const float bi = b2a[col];
    #pragma unroll
    for (int v = 0; v < 8; ++v) {
      float x = acc[j][v] + bi;
      x = (x >= 0.f) ? x : SLOPE * x;
      h1[(size_t)(mt * 16 + half * 8 + v) * FD4 + col] = f2bf(x);
    }
  }
  __syncthreads();

  // ---- GEMM2: [32,512] @ W2b [512,128] -> 16 tiles, 2 per wave ----
  v8f acc2[2];
  acc2[0] = (v8f){}; acc2[1] = (v8f){};

  for (int k = 0; k < FD4; k += 32) {
    const int k0 = k + half * 16;
    Frag16 a;
    a.h[0] = *(const v8bf*)&h1[(size_t)rA * FD4 + k0];
    a.h[1] = *(const v8bf*)&h1[(size_t)rA * FD4 + k0 + 8];
    #pragma unroll
    for (int j = 0; j < 2; ++j) {
      const int n = (ngrp * 2 + j) * 16 + l16;
      const v16bf b = *(const v16bf*)(w2bT + (size_t)n * FD4 + k + half * 16);
      acc2[j] = __builtin_amdgcn_wmma_f32_16x16x32_bf16(
          false, a.v, false, b, (short)0, acc2[j], false, false);
    }
  }

  #pragma unroll
  for (int j = 0; j < 2; ++j) {
    const int col  = (ngrp * 2 + j) * 16 + l16;
    const float bi = b2b[col];
    #pragma unroll
    for (int v = 0; v < 8; ++v) {
      const int tt = t0 + mt * 16 + half * 8 + v;
      if (tt < nT) y[(size_t)tt * FD + col] = acc2[j][v] + bi;
    }
  }
}

// ---------- BatchNorm: column-wise sum / sumsq, then normalize -----------------
__global__ void bn_stats_kernel(const float* __restrict__ y,
                                float* __restrict__ stats, int nT) {
  const int c = threadIdx.x & 127;
  int r = blockIdx.x * 2 + (threadIdx.x >> 7);
  const int stride = gridDim.x * 2;
  float s = 0.f, q = 0.f;
  for (; r < nT; r += stride) {
    const float v = y[(size_t)r * FD + c];
    s += v; q += v * v;
  }
  atomicAdd(&stats[c], s);
  atomicAdd(&stats[FD + c], q);
}

__global__ void bn_apply_kernel(const float* __restrict__ y,
                                const float* __restrict__ stats,
                                const float* __restrict__ gamma,
                                const float* __restrict__ beta,
                                float* __restrict__ out, int nT) {
  const long i = (long)blockIdx.x * blockDim.x + threadIdx.x;
  const long total = (long)nT * FD;
  if (i >= total) return;
  const int c = (int)(i & 127);
  const float inv  = 1.f / (float)nT;
  const float mean = stats[c] * inv;
  const float var  = stats[FD + c] * inv - mean * mean;
  out[i] = (y[i] - mean) * rsqrtf(var + EPSBN) * gamma[c] + beta[c];
}

// ---------------------------------------------------------------------------
extern "C" void kernel_launch(void* const* d_in, const int* in_sizes, int n_in,
                              void* d_out, int out_size, void* d_ws, size_t ws_size,
                              hipStream_t stream) {
  const float* x_s  = (const float*)d_in[0];
  const float* x_t  = (const float*)d_in[1];
  const float* e_at = (const float*)d_in[2];
  const float* u    = (const float*)d_in[3];
  const int*   eidx = (const int*)d_in[4];
  const float* W1a  = (const float*)d_in[5];
  const float* b1a  = (const float*)d_in[6];
  const float* W1b  = (const float*)d_in[7];
  const float* b1b  = (const float*)d_in[8];
  const float* W2a  = (const float*)d_in[9];
  const float* b2a  = (const float*)d_in[10];
  const float* W2b  = (const float*)d_in[11];
  const float* b2b  = (const float*)d_in[12];
  const float* gamma = (const float*)d_in[13];
  const float* beta  = (const float*)d_in[14];

  const int S = in_sizes[0] / FD;
  const int T = in_sizes[1] / FD;
  const int E = in_sizes[4] / 2;
  const int* srcI = eidx;
  const int* tgtI = eidx + E;

  // workspace carve-out (256B aligned)
  char* ws = (char*)d_ws;
  size_t off = 0;
  auto take = [&](size_t bytes) -> char* {
    char* p = ws + off;
    off += (bytes + 255) & ~(size_t)255;
    return p;
  };
  unsigned short* w1aT = (unsigned short*)take((size_t)FD2 * FD2 * 2);
  unsigned short* w1bT = (unsigned short*)take((size_t)FD2 * FD2 * 2);
  unsigned short* w2aT = (unsigned short*)take((size_t)FD4 * FD4 * 2);
  unsigned short* w2bT = (unsigned short*)take((size_t)FD  * FD4 * 2);
  unsigned short* xs_b = (unsigned short*)take((size_t)S * FD * 2);
  unsigned short* ea_b = (unsigned short*)take((size_t)E * FD * 2);
  unsigned short* xt_b = (unsigned short*)take((size_t)T * FD * 2);
  unsigned short* u_b  = (unsigned short*)take((size_t)FD * 2);
  float*          agg  = (float*)take((size_t)T * FD2 * 4);
  unsigned short* aggb = (unsigned short*)take((size_t)T * FD2 * 2);
  float*          yf   = (float*)take((size_t)T * FD * 4);
  float*          stat = (float*)take((size_t)2 * FD * 4);
  (void)ws_size;

  hipMemsetAsync(agg,  0, (size_t)T * FD2 * 4, stream);
  hipMemsetAsync(stat, 0, (size_t)2 * FD * 4, stream);

  const int TB = 256;
  auto blocks = [](long n, int tb) { return (int)((n + tb - 1) / tb); };

  // weights -> bf16 transposed
  cvtT_bf16_kernel<<<blocks((long)FD2 * FD2, TB), TB, 0, stream>>>(W1a, w1aT, FD2, FD2);
  cvtT_bf16_kernel<<<blocks((long)FD2 * FD2, TB), TB, 0, stream>>>(W1b, w1bT, FD2, FD2);
  cvtT_bf16_kernel<<<blocks((long)FD4 * FD4, TB), TB, 0, stream>>>(W2a, w2aT, FD4, FD4);
  cvtT_bf16_kernel<<<blocks((long)FD4 * FD,  TB), TB, 0, stream>>>(W2b, w2bT, FD4, FD);

  // activations -> bf16
  cvt_bf16_kernel<<<blocks((long)S * FD, TB), TB, 0, stream>>>(x_s,  xs_b, (long)S * FD);
  cvt_bf16_kernel<<<blocks((long)E * FD, TB), TB, 0, stream>>>(e_at, ea_b, (long)E * FD);
  cvt_bf16_kernel<<<blocks((long)T * FD, TB), TB, 0, stream>>>(x_t,  xt_b, (long)T * FD);
  cvt_bf16_kernel<<<blocks((long)FD, TB), TB, 0, stream>>>(u, u_b, (long)FD);

  // edge MLP + scatter-sum
  edge_mlp_kernel<<<(E + 63) / 64, 256, 0, stream>>>(
      xs_b, ea_b, w1aT, b1a, w1bT, b1b, srcI, tgtI, agg, E);

  // agg -> bf16 for the node GEMM
  cvt_bf16_kernel<<<blocks((long)T * FD2, TB), TB, 0, stream>>>(agg, aggb, (long)T * FD2);

  // node MLP
  node_mlp_kernel<<<(T + 31) / 32, 256, 0, stream>>>(
      xt_b, aggb, u_b, w2aT, b2a, w2bT, b2b, yf, T);

  // BatchNorm
  bn_stats_kernel<<<256, 256, 0, stream>>>(yf, stat, T);
  bn_apply_kernel<<<blocks((long)T * FD, TB), TB, 0, stream>>>(
      yf, stat, gamma, beta, (float*)d_out, T);
}
